// SoftMaxClassifier_84507776516528
// MI455X (gfx1250) — matrix operand order repaired and checked by reading, not compiled, not run on hardware
//
#include <hip/hip_runtime.h>
#include <hip/hip_bf16.h>
#include <string.h>

// ---------------------------------------------------------------------------
// Problem constants (from reference)
// ---------------------------------------------------------------------------
#define BATCH   16
#define RROWS   4096
#define DDIM    1024
#define EDIM    128
#define KCLS    21          // classes incl. background
#define KKNOWN  20          // foreground classes
#define NPAD    32          // padded class dim for two 16-wide WMMA N tiles
#define PRENMS  1024
#define TOPK    100
#define STHR    0.05f
#define NMSTHR  0.5f

typedef __attribute__((ext_vector_type(16))) __bf16 v16bf;
typedef __attribute__((ext_vector_type(8)))  float  v8f;

union VB16 { v16bf v; __bf16 e[16]; };

// ---------------------------------------------------------------------------
// Workspace layout (bytes)
// ---------------------------------------------------------------------------
#define WS_PROBS_OFF   ((size_t)0)                               // 16*4096*20 f32
#define WS_BOXC_OFF    ((size_t)(BATCH * RROWS * KKNOWN) * 4)    // 16*4096*4 f32
#define WS_WHI_OFF     (WS_BOXC_OFF + (size_t)(BATCH * RROWS * 4) * 4)
#define WS_WLO_OFF     (WS_WHI_OFF + (size_t)(NPAD * DDIM) * 2)

// ---------------------------------------------------------------------------
// Kernel 1: split classifier weights into padded bf16 hi/lo planes.
// W: [21][1024] f32 -> Whi/Wlo: [32][1024] bf16 (rows 21..31 zero)
// ---------------------------------------------------------------------------
__global__ void wprep_kernel(const float* __restrict__ W,
                             __bf16* __restrict__ whi,
                             __bf16* __restrict__ wlo) {
    int i = blockIdx.x * blockDim.x + threadIdx.x;
    int stride = gridDim.x * blockDim.x;
    for (; i < NPAD * DDIM; i += stride) {
        int n = i >> 10;
        float f = (n < KCLS) ? W[n * DDIM + (i & (DDIM - 1))] : 0.0f;
        __bf16 h = (__bf16)f;
        whi[i] = h;
        wlo[i] = (__bf16)(f - (float)h);
    }
}

// ---------------------------------------------------------------------------
// Kernel 2: WMMA GEMM (logits) + softmax (drop background) + box clip.
// Memory-bound (268 MB of features @ 23.3 TB/s ~= 11.5 us floor); bf16
// split-product (Ahi*Bhi + Alo*Bhi + Ahi*Blo) recovers ~f32 accuracy on the
// WMMA pipe at negligible cost. Block = 256 threads = 8 waves; each wave
// computes a 16-row x 32-padded-class tile.
// ---------------------------------------------------------------------------
__global__ __launch_bounds__(256) void gemm_softmax_kernel(
    const float* __restrict__ features,   // [B*R][1024]
    const float* __restrict__ boxes,      // [B*R][4]
    const __bf16* __restrict__ whi,       // [32][1024]
    const __bf16* __restrict__ wlo,       // [32][1024]
    const float* __restrict__ bias,       // [21]
    const int* __restrict__ ph, const int* __restrict__ pw,
    float* __restrict__ probs,            // [B*R][20]
    float* __restrict__ boxesc)           // [B*R][4]
{
    __shared__ float sLogit[128][NPAD + 1];   // +1 pad vs bank conflicts

    const int tid  = threadIdx.x;
    const int wave = tid >> 5;
    const int lane = tid & 31;
    const int half = lane >> 4;      // K-half selector per WMMA layout
    const int m    = lane & 15;      // row within tile / col within N tile

    const int row0 = blockIdx.x * 128 + wave * 16;
    const float* aRow = features + (size_t)(row0 + m) * DDIM;

    const __bf16* bh0 = whi + (size_t)m * DDIM;
    const __bf16* bl0 = wlo + (size_t)m * DDIM;
    const __bf16* bh1 = whi + (size_t)(16 + m) * DDIM;
    const __bf16* bl1 = wlo + (size_t)(16 + m) * DDIM;

    v8f acc0 = {};
    v8f acc1 = {};

    for (int k0 = 0; k0 < DDIM; k0 += 32) {
        // A fragment: element e -> K = k0+half*16+e
        const float4 f0 = *(const float4*)(aRow + k0 + half * 16);
        const float4 f1 = *(const float4*)(aRow + k0 + half * 16 + 4);
        const float4 f2 = *(const float4*)(aRow + k0 + half * 16 + 8);
        const float4 f3 = *(const float4*)(aRow + k0 + half * 16 + 12);
        if (k0 + 32 < DDIM)
            __builtin_prefetch(aRow + k0 + 32 + half * 8, 0, 1);  // global_prefetch_b8

        float tmp[16] = { f0.x, f0.y, f0.z, f0.w, f1.x, f1.y, f1.z, f1.w,
                          f2.x, f2.y, f2.z, f2.w, f3.x, f3.y, f3.z, f3.w };
        VB16 ahi, alo;
#pragma unroll
        for (int e = 0; e < 16; ++e) {
            __bf16 h = (__bf16)tmp[e];
            ahi.e[e] = h;
            alo.e[e] = (__bf16)(tmp[e] - (float)h);
        }

        // B fragment: lane L -> N = L%16 (+16 for tile 1), element e -> K = k0 + (L/16)*16 + e
        const int koff = k0 + half * 16;
        const v16bf bh0v = *(const v16bf*)(bh0 + koff);
        const v16bf bl0v = *(const v16bf*)(bl0 + koff);
        const v16bf bh1v = *(const v16bf*)(bh1 + koff);
        const v16bf bl1v = *(const v16bf*)(bl1 + koff);

        acc0 = __builtin_amdgcn_wmma_f32_16x16x32_bf16(false, ahi.v, false, bh0v, (short)0, acc0, false, false);
        acc0 = __builtin_amdgcn_wmma_f32_16x16x32_bf16(false, alo.v, false, bh0v, (short)0, acc0, false, false);
        acc0 = __builtin_amdgcn_wmma_f32_16x16x32_bf16(false, ahi.v, false, bl0v, (short)0, acc0, false, false);
        acc1 = __builtin_amdgcn_wmma_f32_16x16x32_bf16(false, ahi.v, false, bh1v, (short)0, acc1, false, false);
        acc1 = __builtin_amdgcn_wmma_f32_16x16x32_bf16(false, alo.v, false, bh1v, (short)0, acc1, false, false);
        acc1 = __builtin_amdgcn_wmma_f32_16x16x32_bf16(false, ahi.v, false, bl1v, (short)0, acc1, false, false);
    }

    // C/D layout: VGPR v, lane L -> M = v + 8*(L/16), N = L%16
#pragma unroll
    for (int v = 0; v < 8; ++v) {
        int rl = wave * 16 + v + 8 * half;
        sLogit[rl][m]      = acc0[v];
        sLogit[rl][16 + m] = acc1[v];
    }
    __syncthreads();

    const int g128 = blockIdx.x * 128;
    if (tid < 128) {
        // softmax over 21 real classes, drop background column
        float l[KCLS];
        float mx = -1e30f;
#pragma unroll
        for (int c = 0; c < KCLS; ++c) {
            l[c] = sLogit[tid][c] + bias[c];
            mx = fmaxf(mx, l[c]);
        }
        float sum = 0.0f;
#pragma unroll
        for (int c = 0; c < KCLS; ++c) { l[c] = __expf(l[c] - mx); sum += l[c]; }
        const float inv = 1.0f / sum;
        const int g = g128 + tid;
#pragma unroll
        for (int c = 0; c < KKNOWN; ++c)
            probs[(size_t)g * KKNOWN + c] = l[c] * inv;
    } else {
        // box clipping for one row
        const int g = g128 + (tid - 128);
        const float w = (float)(*pw), h = (float)(*ph);
        const float* bp = boxes + (size_t)g * 4;
        float* o = boxesc + (size_t)g * 4;
        o[0] = fminf(fmaxf(bp[0], 0.0f), w);
        o[1] = fminf(fmaxf(bp[1], 0.0f), h);
        o[2] = fminf(fmaxf(bp[2], 0.0f), w);
        o[3] = fminf(fmaxf(bp[3], 0.0f), h);
    }
}

// inclusive Hillis-Steele scan over 1024 ints in LDS; all 1024 threads call.
__device__ inline int block_scan1024(int* buf, int tid) {
    for (int off = 1; off < 1024; off <<= 1) {
        int v = (tid >= off) ? buf[tid - off] : 0;
        __syncthreads();
        buf[tid] += v;
        __syncthreads();
    }
    return buf[tid];
}

// ---------------------------------------------------------------------------
// Kernel 3: per-image threshold -> deterministic top-1024 -> bitonic sort ->
// mask-based greedy NMS (128 KB dynamic LDS suppression matrix, single-wave
// greedy reduction, no division, no per-iteration block barriers) ->
// top-100 emit + embedding gather. One block (1024 threads = 32 waves)/image.
// ---------------------------------------------------------------------------
__global__ __launch_bounds__(1024) void select_nms_kernel(
    const float* __restrict__ probs,    // [B][R][20]
    const float* __restrict__ boxesc,   // [B][R][4]
    const float* __restrict__ emb,      // [B][R][128]
    float* __restrict__ out)
{
    extern __shared__ unsigned sMask[];  // [PRENMS][32] suppression bitmask (128 KB dynamic)

    __shared__ int      sHist[4096];     // histogram, then reused as scan buffer
    __shared__ float    sScore[PRENMS];
    __shared__ int      sCode[PRENMS];   // flat index r*20+c
    __shared__ float    sBX1[PRENMS], sBY1[PRENMS], sBX2[PRENMS], sBY2[PRENMS];
    __shared__ float    sArea[PRENMS];
    __shared__ int      sCls[PRENMS];
    __shared__ unsigned sKeepWord[32];   // final keep bitmask (1 bit/candidate)
    __shared__ int      sOutCode[TOPK];
    __shared__ int      sT;

    const int tid = threadIdx.x;
    const int img = blockIdx.x;
    const int NPE = RROWS * KKNOWN;          // 81920
    const int CHK = NPE / 1024;              // 80 entries per thread (contiguous)
    const float* p = probs + (size_t)img * NPE;

    // init
    sHist[tid] = 0; sHist[tid + 1024] = 0; sHist[tid + 2048] = 0; sHist[tid + 3072] = 0;
    sScore[tid] = -1.0f; sCode[tid] = 0;
    __syncthreads();

    const int base = tid * CHK;

    // --- histogram of thresholded scores (4096 bins over (0,1)) ---
    for (int q = 0; q < CHK; ++q) {
        float s = p[base + q];
        if (s > STHR) {
            int bin = (int)(s * 4096.0f); if (bin > 4095) bin = 4095;
            atomicAdd(&sHist[bin], 1);
        }
    }
    __syncthreads();

    // --- cutoff bin T: bins > T hold < 1024 entries, bins >= T hold >= 1024 ---
    if (tid == 0) {
        int acc = 0, chosen = 0;
        for (int b = 4095; b >= 0; --b) {
            acc += sHist[b];
            if (acc >= PRENMS) { chosen = b; break; }
        }
        sT = chosen;
    }
    __syncthreads();
    const int T = sT;

    // --- pass A: all entries with bin > T, deterministic rank via chunk scan ---
    int cntA = 0;
    for (int q = 0; q < CHK; ++q) {
        float s = p[base + q];
        if (s > STHR) {
            int bin = (int)(s * 4096.0f); if (bin > 4095) bin = 4095;
            if (bin > T) cntA++;
        }
    }
    __syncthreads();
    sHist[tid] = cntA;
    __syncthreads();
    int inclA = block_scan1024(sHist, tid);
    int wpos = inclA - cntA;
    const int totalA = sHist[1023];
    for (int q = 0; q < CHK; ++q) {
        float s = p[base + q];
        if (s > STHR) {
            int bin = (int)(s * 4096.0f); if (bin > 4095) bin = 4095;
            if (bin > T) { sScore[wpos] = s; sCode[wpos] = base + q; wpos++; }
        }
    }
    __syncthreads();

    // --- pass B: fill remainder from bin == T (capped, deterministic order) ---
    int cntB = 0;
    for (int q = 0; q < CHK; ++q) {
        float s = p[base + q];
        if (s > STHR) {
            int bin = (int)(s * 4096.0f); if (bin > 4095) bin = 4095;
            if (bin == T) cntB++;
        }
    }
    __syncthreads();
    sHist[tid] = cntB;
    __syncthreads();
    int inclB = block_scan1024(sHist, tid);
    wpos = totalA + inclB - cntB;
    for (int q = 0; q < CHK; ++q) {
        float s = p[base + q];
        if (s > STHR) {
            int bin = (int)(s * 4096.0f); if (bin > 4095) bin = 4095;
            if (bin == T) {
                if (wpos < PRENMS) { sScore[wpos] = s; sCode[wpos] = base + q; }
                wpos++;
            }
        }
    }
    __syncthreads();

    // --- bitonic sort (descending by score) on (score, code) ---
    for (int k = 2; k <= PRENMS; k <<= 1) {
        for (int j = k >> 1; j > 0; j >>= 1) {
            int ixj = tid ^ j;
            if (ixj > tid) {
                bool up = ((tid & k) == 0);
                float s1 = sScore[tid], s2 = sScore[ixj];
                bool sw = up ? (s1 < s2) : (s1 > s2);
                if (sw) {
                    sScore[tid] = s2; sScore[ixj] = s1;
                    int c1 = sCode[tid]; sCode[tid] = sCode[ixj]; sCode[ixj] = c1;
                }
            }
            __syncthreads();
        }
    }

    // --- candidate geometry (SoA for LDS-broadcast reads) ---
    {
        const int code = sCode[tid];
        const float* bp = boxesc + ((size_t)img * RROWS + (code / KKNOWN)) * 4;
        const float b0 = bp[0], b1 = bp[1], b2 = bp[2], b3 = bp[3];
        sBX1[tid] = b0; sBY1[tid] = b1; sBX2[tid] = b2; sBY2[tid] = b3;
        sArea[tid] = fmaxf(b2 - b0, 0.0f) * fmaxf(b3 - b1, 0.0f);
        sCls[tid]  = code % KKNOWN;     // same-class test == class-offset trick
    }
    __syncthreads();

    // --- suppression bitmask: thread tid owns row i=tid; inner j-loop is
    //     uniform across the wave so every sXX[j] read is an LDS broadcast.
    //     Division-free: inter/union > 0.5  <=>  3*inter > areaI + areaJ. ---
    {
        const float x1 = sBX1[tid], y1 = sBY1[tid];
        const float x2 = sBX2[tid], y2 = sBY2[tid];
        const float ar = sArea[tid];
        const int   cl = sCls[tid];
        for (int w = 0; w < 32; ++w) {
            unsigned bits = 0u;
#pragma unroll 8
            for (int b = 0; b < 32; ++b) {
                const int j = w * 32 + b;
                if (j > tid && sCls[j] == cl) {
                    const float iw = fmaxf(fminf(x2, sBX2[j]) - fmaxf(x1, sBX1[j]), 0.0f);
                    const float ih = fmaxf(fminf(y2, sBY2[j]) - fmaxf(y1, sBY1[j]), 0.0f);
                    const float inter = iw * ih;
                    if (3.0f * inter > ar + sArea[j]) bits |= 1u << b;
                }
            }
            sMask[tid * 32 + w] = bits;
        }
    }
    __syncthreads();

    // --- single-wave greedy reduction: lane l owns removal word l.
    //     1024 iterations of shfl + ds_load + or; zero block barriers. ---
    if (tid < 32) {
        unsigned removed = 0u;
#pragma unroll
        for (int b = 0; b < 32; ++b)
            removed |= (sScore[tid * 32 + b] > STHR ? 0u : 1u) << b;   // invalid = pre-removed
        for (int i = 0; i < PRENMS; ++i) {
            const unsigned ri = __shfl(removed, i >> 5);
            if (!((ri >> (i & 31)) & 1u))           // candidate i survives -> apply its row
                removed |= sMask[i * 32 + tid];
        }
        sKeepWord[tid] = ~removed;
    }
    __syncthreads();

    const int keep = (sKeepWord[tid >> 5] >> (tid & 31)) & 1;

    // --- keep prefix scan -> output slots (kept first in order, then fillers) ---
    sHist[tid] = keep;
    __syncthreads();
    int incl = block_scan1024(sHist, tid);
    const int nkept = sHist[1023];
    const int excl = incl - keep;
    const int slot = keep ? excl : (nkept + (tid - incl));

    float* oBoxes  = out;                                     // [16][100][4]
    float* oScores = out + BATCH * TOPK * 4;                  // [16][100]
    float* oCls    = out + BATCH * TOPK * 5;                  // [16][100]
    float* oEmb    = out + BATCH * TOPK * 6;                  // [16][100][128]
    float* oValid  = out + BATCH * TOPK * 6 + BATCH * TOPK * EDIM;

    if (slot < TOPK) {
        const int code = sCode[tid];
        const int cls  = code % KKNOWN;
        const size_t o4 = ((size_t)img * TOPK + slot) * 4;
        oBoxes[o4 + 0] = sBX1[tid]; oBoxes[o4 + 1] = sBY1[tid];
        oBoxes[o4 + 2] = sBX2[tid]; oBoxes[o4 + 3] = sBY2[tid];
        oScores[img * TOPK + slot] = keep ? sScore[tid] : -1.0f;
        oCls[img * TOPK + slot]    = (float)cls;
        oValid[img * TOPK + slot]  = keep ? 1.0f : 0.0f;
        sOutCode[slot] = code;
    }
    __syncthreads();

    // --- cooperative embedding gather for the 100 selected ---
    const float* eimg = emb + (size_t)img * RROWS * EDIM;
    for (int e2 = tid; e2 < TOPK * EDIM; e2 += 1024) {
        const int slot2 = e2 >> 7;
        const int ln    = e2 & (EDIM - 1);
        const int code  = sOutCode[slot2];
        oEmb[((size_t)img * TOPK + slot2) * EDIM + ln] =
            eimg[(size_t)(code / KKNOWN) * EDIM + ln];
    }
}

// ---------------------------------------------------------------------------
// Host launcher
// ---------------------------------------------------------------------------
extern "C" void kernel_launch(void* const* d_in, const int* in_sizes, int n_in,
                              void* d_out, int out_size, void* d_ws, size_t ws_size,
                              hipStream_t stream) {
    (void)in_sizes; (void)n_in; (void)out_size; (void)ws_size;

    const float* features = (const float*)d_in[0];   // [16,4096,1024]
    const float* boxes    = (const float*)d_in[1];   // [16,4096,4]
    const float* pemb     = (const float*)d_in[2];   // [16,4096,128]
    const float* W        = (const float*)d_in[3];   // [21,1024]
    const float* bvec     = (const float*)d_in[4];   // [21]
    const int*   ph       = (const int*)d_in[5];     // scalar H
    const int*   pw       = (const int*)d_in[6];     // scalar W
    float* out = (float*)d_out;

    char* ws = (char*)d_ws;
    float*  probs  = (float*)(ws + WS_PROBS_OFF);
    float*  boxesc = (float*)(ws + WS_BOXC_OFF);
    __bf16* whi    = (__bf16*)(ws + WS_WHI_OFF);
    __bf16* wlo    = (__bf16*)(ws + WS_WLO_OFF);

    // 1) weight split to padded bf16 hi/lo planes
    wprep_kernel<<<32, 256, 0, stream>>>(W, whi, wlo);

    // 2) WMMA GEMM + softmax + box clip: 65536 rows, 128 rows/block
    gemm_softmax_kernel<<<(BATCH * RROWS) / 128, 256, 0, stream>>>(
        features, boxes, whi, wlo, bvec, ph, pw, probs, boxesc);

    // 3) per-image selection + mask-NMS + top-100 emit
    //    128 KB dynamic LDS suppression matrix (CDNA5: 320 KB LDS per WGP)
    select_nms_kernel<<<BATCH, 1024, PRENMS * 32 * sizeof(unsigned), stream>>>(
        probs, boxesc, pemb, out);
}
